// GNN2D_24146306138677
// MI455X (gfx1250) — matrix operand order repaired and checked by reading, not compiled, not run on hardware
//
#include <hip/hip_runtime.h>
#include <hip/hip_bf16.h>

typedef __bf16 bf16_t;
typedef __attribute__((ext_vector_type(16))) __bf16 v16bf;
typedef __attribute__((ext_vector_type(8)))  float  v8f;

#define HID      128
#define NNODES   400000
#define NEDGES   1600000
#define NMOLS    20000
#define MROWS    64      // node rows per WMMA block
#define PITCH    136     // LDS row pitch (bf16 elems), 272B = 16B aligned, bank-skewed

union FragAB { v16bf v; uint4 q[2]; };
union Pack4  { bf16_t h[4]; uint2 u; };

__device__ __forceinline__ float silu_f(float x) { return x / (1.0f + __expf(-x)); }

// ---------------------------------------------------------------------------
// Convert fp32 [K=128][N=128] row-major weight -> bf16 [N][K] (transposed)
// ---------------------------------------------------------------------------
__global__ __launch_bounds__(256) void conv_wt(const float* __restrict__ src,
                                               bf16_t* __restrict__ dst) {
  int t = blockIdx.x * 256 + threadIdx.x;   // 0..16383
  int n = t >> 7;
  int k = t & 127;
  dst[n * HID + k] = (bf16_t)src[k * HID + n];
}

__global__ __launch_bounds__(256) void zero_k(float4* __restrict__ p, int n4) {
  int i = blockIdx.x * 256 + threadIdx.x;
  if (i < n4) p[i] = make_float4(0.f, 0.f, 0.f, 0.f);
}

// ---------------------------------------------------------------------------
// h = x @ atom_W + atom_b     (block = 256 thr = 2 nodes)
// ---------------------------------------------------------------------------
__global__ __launch_bounds__(256) void atom_embed(const float* __restrict__ x,
                                                  const float* __restrict__ W,
                                                  const float* __restrict__ b,
                                                  float* __restrict__ h) {
  __shared__ float xs[2][56];
  const int ln = threadIdx.x >> 7, j = threadIdx.x & 127;
  const long n = (long)blockIdx.x * 2 + ln;
  if (j < 53) xs[ln][j] = x[n * 53 + j];
  __syncthreads();
  float acc = b[j];
  #pragma unroll
  for (int k = 0; k < 53; ++k) acc += xs[ln][k] * W[k * HID + j];
  h[n * HID + j] = acc;
}

// ---------------------------------------------------------------------------
// Per-edge message: m = silu(h[src] + edge_attr@We + be); agg[dst] += m
// (edge embedding recomputed on the fly: saves an 819MB buffer + its traffic)
// ---------------------------------------------------------------------------
__global__ __launch_bounds__(256) void edge_msg(const float* __restrict__ h,
                                                const float* __restrict__ eattr,
                                                const int* __restrict__ eidx,
                                                const float* __restrict__ eW,
                                                const float* __restrict__ eb,
                                                float* __restrict__ agg) {
  __shared__ float ea[2][16];
  const int le = threadIdx.x >> 7, j = threadIdx.x & 127;
  const long e = (long)blockIdx.x * 2 + le;
  if (j < 14) ea[le][j] = eattr[e * 14 + j];
  __syncthreads();
  const int src = eidx[e];
  const int dst = eidx[(long)NEDGES + e];
  float acc = eb[j];
  #pragma unroll
  for (int k = 0; k < 14; ++k) acc += ea[le][k] * eW[k * HID + j];
  float v = h[(long)src * HID + j] + acc;
  v = silu_f(v);
  atomicAdd(&agg[(long)dst * HID + j], v);
}

// ---------------------------------------------------------------------------
// WMMA node MLP.
//   MODE 0: z=(1+eps)h+agg; h' = silu(silu(z@W1+b1)@W2+b2)   (GIN layer)
//   MODE 1: h' = h@W1+b1                                     (output proj)
// Block = 128 threads (4 wave32), 64 rows. Each wave owns a private 16-row
// strip, so no cross-wave LDS dependencies. Weights pre-transposed bf16 [N][K].
// ---------------------------------------------------------------------------
template <int MODE>
__global__ __launch_bounds__(128) void gin_mlp(
    const float* __restrict__ hIn, const float* __restrict__ agg,
    const float* __restrict__ epsPtr,
    const bf16_t* __restrict__ w1t, const float* __restrict__ b1,
    const bf16_t* __restrict__ w2t, const float* __restrict__ b2,
    float* __restrict__ hOut) {
  __shared__ bf16_t lA[MROWS * PITCH];
  __shared__ bf16_t lB[MROWS * PITCH];

  const int tid  = threadIdx.x;
  const int wave = tid >> 5;
  const int lane = tid & 31;
  const long rowBase = (long)blockIdx.x * MROWS;

  if (tid == 0) {
    __builtin_prefetch(w1t, 0, 1);
    if (MODE == 0) __builtin_prefetch(w2t, 0, 1);
  }

  // ---- Stage 0: build bf16 A tile in LDS (each wave writes its own 16 rows)
  {
    const int r  = tid >> 1;            // 0..63
    const int c0 = (tid & 1) * 64;
    const float4* hp = (const float4*)(hIn + (rowBase + r) * HID + c0);
    const float4* ap = nullptr;
    float eps1 = 1.0f;
    if (MODE == 0) {
      eps1 = 1.0f + epsPtr[0];
      ap = (const float4*)(agg + (rowBase + r) * HID + c0);
    }
    uint2* dst = (uint2*)(lA + r * PITCH + c0);
    #pragma unroll
    for (int i = 0; i < 16; ++i) {
      float4 v = hp[i];
      if (MODE == 0) {
        float4 a = ap[i];
        v.x = eps1 * v.x + a.x; v.y = eps1 * v.y + a.y;
        v.z = eps1 * v.z + a.z; v.w = eps1 * v.w + a.w;
      }
      Pack4 p;
      p.h[0] = (bf16_t)v.x; p.h[1] = (bf16_t)v.y;
      p.h[2] = (bf16_t)v.z; p.h[3] = (bf16_t)v.w;
      dst[i] = p.u;
    }
  }
  __syncthreads();

  const int half = lane >> 4;   // K-half selector per ISA A/B layouts
  const int lr   = lane & 15;   // A: row-in-tile, B/C/D: col-in-tile
  const int rA   = wave * 16;

  // ---- GEMM 1:  [16x128] x [128x128]
  {
    FragAB a[4];
    #pragma unroll
    for (int kt = 0; kt < 4; ++kt) {   // A 16x32 tile, contiguous k = half*16 + s, same order as B
      const bf16_t* base = lA + (rA + lr) * PITCH + kt * 32 + half * 16;
      a[kt].q[0] = *(const uint4*)(base);        // K = half*16  .. +7
      a[kt].q[1] = *(const uint4*)(base + 8);    // K = half*16+8.. +7
    }
    #pragma unroll
    for (int nt = 0; nt < 8; ++nt) {
      const float bv = b1[nt * 16 + lr];
      v8f acc;
      #pragma unroll
      for (int i = 0; i < 8; ++i) acc[i] = bv;
      #pragma unroll
      for (int kt = 0; kt < 4; ++kt) {
        FragAB bfr;                      // B col = lane, K contiguous (transposed W)
        const bf16_t* bp = w1t + (nt * 16 + lr) * HID + kt * 32 + half * 16;
        bfr.q[0] = ((const uint4*)bp)[0];
        bfr.q[1] = ((const uint4*)bp)[1];
        acc = __builtin_amdgcn_wmma_f32_16x16x32_bf16(false, a[kt].v, false, bfr.v,
                                                      (short)0, acc, false, false);
      }
      if (MODE == 0) {                   // D layout: VGPR i -> row half*8+i, col lr
        #pragma unroll
        for (int i = 0; i < 8; ++i)
          lB[(rA + half * 8 + i) * PITCH + nt * 16 + lr] = (bf16_t)silu_f(acc[i]);
      } else {
        #pragma unroll
        for (int i = 0; i < 8; ++i)
          hOut[(rowBase + rA + half * 8 + i) * HID + nt * 16 + lr] = acc[i];
      }
    }
  }

  if (MODE == 0) {
    __syncthreads();
    // ---- GEMM 2 on silu(GEMM1) from LDS
    FragAB a[4];
    #pragma unroll
    for (int kt = 0; kt < 4; ++kt) {
      const bf16_t* base = lB + (rA + lr) * PITCH + kt * 32 + half * 16;
      a[kt].q[0] = *(const uint4*)(base);
      a[kt].q[1] = *(const uint4*)(base + 8);
    }
    #pragma unroll
    for (int nt = 0; nt < 8; ++nt) {
      const float bv = b2[nt * 16 + lr];
      v8f acc;
      #pragma unroll
      for (int i = 0; i < 8; ++i) acc[i] = bv;
      #pragma unroll
      for (int kt = 0; kt < 4; ++kt) {
        FragAB bfr;
        const bf16_t* bp = w2t + (nt * 16 + lr) * HID + kt * 32 + half * 16;
        bfr.q[0] = ((const uint4*)bp)[0];
        bfr.q[1] = ((const uint4*)bp)[1];
        acc = __builtin_amdgcn_wmma_f32_16x16x32_bf16(false, a[kt].v, false, bfr.v,
                                                      (short)0, acc, false, false);
      }
      #pragma unroll
      for (int i = 0; i < 8; ++i)
        hOut[(rowBase + rA + half * 8 + i) * HID + nt * 16 + lr] = silu_f(acc[i]);
    }
  }
}

// ---------------------------------------------------------------------------
// h_agg[batch[n]] += hOut[n]     (batch sorted; f32 atomics into 10MB buffer)
// ---------------------------------------------------------------------------
__global__ __launch_bounds__(256) void segsum(const float* __restrict__ hOut,
                                              const int* __restrict__ batch,
                                              float* __restrict__ hAgg) {
  const long i = (long)blockIdx.x * 256 + threadIdx.x;   // 0..51.2M-1
  const int n = (int)(i >> 7);
  const int c = (int)(i & 127);
  atomicAdd(&hAgg[(long)batch[n] * HID + c], hOut[i]);
}

// ---------------------------------------------------------------------------
// Readout head: 1 block (128 thr) per molecule.
// h_bond = pw(a,b)+pw(b,a); out = pp(concat(h_bond, h_agg))
// ---------------------------------------------------------------------------
__global__ __launch_bounds__(128) void head_kernel(
    const float* __restrict__ hOut, const int* __restrict__ selS,
    const int* __restrict__ selE, const float* __restrict__ hAgg,
    const float* __restrict__ pwW1, const float* __restrict__ pwb1,
    const float* __restrict__ pwW2, const float* __restrict__ pwb2,
    const float* __restrict__ ppW1, const float* __restrict__ ppb1,
    const float* __restrict__ ppW2, const float* __restrict__ ppb2,
    const float* __restrict__ ppW3, const float* __restrict__ ppb3,
    float* __restrict__ out) {
  __shared__ float sa[128], sb[128], st[128], shb[128], sr[128];
  const int j = threadIdx.x;
  const int m = blockIdx.x;
  sa[j] = hOut[(long)selS[m] * HID + j];
  sb[j] = hOut[(long)selE[m] * HID + j];
  shb[j] = 0.f;
  __syncthreads();

  #pragma unroll
  for (int dir = 0; dir < 2; ++dir) {
    const float* u0 = dir ? sb : sa;
    const float* u1 = dir ? sa : sb;
    float acc = pwb1[j];
    for (int k = 0; k < 128; ++k) acc += u0[k] * pwW1[k * HID + j];
    for (int k = 0; k < 128; ++k) acc += u1[k] * pwW1[(k + 128) * HID + j];
    __syncthreads();
    st[j] = silu_f(acc);
    __syncthreads();
    float acc2 = pwb2[j];
    for (int k = 0; k < 128; ++k) acc2 += st[k] * pwW2[k * HID + j];
    shb[j] += acc2;
    __syncthreads();
  }

  float acc = ppb1[j];
  for (int k = 0; k < 128; ++k) acc += shb[k] * ppW1[k * HID + j];
  const float* hag = hAgg + (long)m * HID;
  for (int k = 0; k < 128; ++k) acc += hag[k] * ppW1[(k + 128) * HID + j];
  __syncthreads();
  st[j] = silu_f(acc);
  __syncthreads();
  float acc2 = ppb2[j];
  for (int k = 0; k < 128; ++k) acc2 += st[k] * ppW2[k * HID + j];
  sr[j] = silu_f(acc2) * ppW3[j];
  __syncthreads();
  if (j == 0) {
    float s = ppb3[0];
    for (int k = 0; k < 128; ++k) s += sr[k];
    out[m] = s;
  }
}

// ---------------------------------------------------------------------------
extern "C" void kernel_launch(void* const* d_in, const int* in_sizes, int n_in,
                              void* d_out, int out_size, void* d_ws, size_t ws_size,
                              hipStream_t stream) {
  (void)in_sizes; (void)n_in; (void)out_size; (void)ws_size;

  const float* x     = (const float*)d_in[0];
  const int*   eidx  = (const int*)d_in[1];
  const float* eattr = (const float*)d_in[2];
  const int*   batch = (const int*)d_in[3];
  const int*   selS  = (const int*)d_in[4];
  const int*   selE  = (const int*)d_in[5];
  const float* atomW = (const float*)d_in[6];
  const float* atomB = (const float*)d_in[7];
  const float* edgeW = (const float*)d_in[8];
  const float* edgeB = (const float*)d_in[9];
  // layers: eps=10+5l, W1=11+5l, b1=12+5l, W2=13+5l, b2=14+5l
  const float* outB  = (const float*)d_in[31];
  const float* pwW1  = (const float*)d_in[32];
  const float* pwb1  = (const float*)d_in[33];
  const float* pwW2  = (const float*)d_in[34];
  const float* pwb2  = (const float*)d_in[35];
  const float* ppW1  = (const float*)d_in[36];
  const float* ppb1  = (const float*)d_in[37];
  const float* ppW2  = (const float*)d_in[38];
  const float* ppb2  = (const float*)d_in[39];
  const float* ppW3  = (const float*)d_in[40];
  const float* ppb3  = (const float*)d_in[41];

  float* out_head = (float*)d_out;               // [20000]
  float* hOut     = (float*)d_out + NMOLS;       // [400000*128]

  char* ws = (char*)d_ws;
  float*  h    = (float*)(ws);                    // 204,800,000 B
  float*  agg  = (float*)(ws + 204800000);        // 204,800,000 B
  float*  hAgg = (float*)(ws + 409600000);        //  10,240,000 B
  bf16_t* wb   = (bf16_t*)(ws + 419840000);       // 9 * 32 KB bf16 weights

  // bf16 transposed weight copies (W1_l -> slot 2l, W2_l -> 2l+1, out_W -> 8)
  for (int l = 0; l < 4; ++l) {
    conv_wt<<<64, 256, 0, stream>>>((const float*)d_in[11 + 5 * l], wb + (2 * l) * 16384);
    conv_wt<<<64, 256, 0, stream>>>((const float*)d_in[13 + 5 * l], wb + (2 * l + 1) * 16384);
  }
  conv_wt<<<64, 256, 0, stream>>>((const float*)d_in[30], wb + 8 * 16384);

  atom_embed<<<NNODES / 2, 256, 0, stream>>>(x, atomW, atomB, h);

  for (int l = 0; l < 4; ++l) {
    zero_k<<<(NNODES * HID / 4) / 256, 256, 0, stream>>>((float4*)agg, NNODES * HID / 4);
    edge_msg<<<NEDGES / 2, 256, 0, stream>>>(h, eattr, eidx, edgeW, edgeB, agg);
    gin_mlp<0><<<NNODES / MROWS, 128, 0, stream>>>(
        h, agg, (const float*)d_in[10 + 5 * l],
        wb + (2 * l) * 16384, (const float*)d_in[12 + 5 * l],
        wb + (2 * l + 1) * 16384, (const float*)d_in[14 + 5 * l], h);
  }

  gin_mlp<1><<<NNODES / MROWS, 128, 0, stream>>>(
      h, nullptr, nullptr, wb + 8 * 16384, outB, nullptr, nullptr, hOut);

  zero_k<<<(NMOLS * HID / 4) / 256, 256, 0, stream>>>((float4*)hAgg, NMOLS * HID / 4);
  segsum<<<(NNODES * HID) / 256, 256, 0, stream>>>(hOut, batch, hAgg);

  head_kernel<<<NMOLS, 128, 0, stream>>>(hOut, selS, selE, hAgg,
                                         pwW1, pwb1, pwW2, pwb2,
                                         ppW1, ppb1, ppW2, ppb2,
                                         ppW3, ppb3, out_head);
}
